// MambaBlock_90383291777613
// MI455X (gfx1250) — matrix operand order repaired and checked by reading, not compiled, not run on hardware
//
#include <hip/hip_runtime.h>

// ---------------------------------------------------------------------------
// Mamba block forward for MI455X (gfx1250).
// bf16 WMMA GEMMs; in_proj staged via Tensor Data Mover (TDM) double-buffered
// LDS tiles, out_proj via register-staged LDS tiles; lane-parallel fp32 scan.
// B=2, L=2048, D_MODEL=1024, D_INNER=2048, DT_RANK=64, D_STATE=16, D_CONV=4
// ---------------------------------------------------------------------------

#define BB      2
#define LL      2048
#define DMODEL  1024
#define DINNER  2048
#define DTRANK  64
#define DSTATE  16
#define DCONV   4
#define NTOK    (BB * LL)          // 4096 tokens

typedef __bf16 v8bf  __attribute__((ext_vector_type(8)));
typedef __bf16 v16bf __attribute__((ext_vector_type(16)));
typedef float  v8f   __attribute__((ext_vector_type(8)));
typedef unsigned int  v4u  __attribute__((ext_vector_type(4)));
typedef int           v4i  __attribute__((ext_vector_type(4)));
typedef int           v8i  __attribute__((ext_vector_type(8)));
typedef unsigned short u16;

// ----------------------------- bf16 helpers --------------------------------
__device__ __forceinline__ u16 f2bf(float f) {
    union { float f; unsigned u; } c; c.f = f;
    unsigned u = c.u;
    u += 0x7FFFu + ((u >> 16) & 1u);          // round-to-nearest-even
    return (u16)(u >> 16);
}
__device__ __forceinline__ float bf2f(u16 b) {
    union { unsigned u; float f; } c; c.u = ((unsigned)b) << 16;
    return c.f;
}

__device__ __forceinline__ v8f wmma_bf16(v16bf a, v16bf b, v8f c) {
    return __builtin_amdgcn_wmma_f32_16x16x32_bf16(false, a, false, b,
                                                   (short)0, c, false, false);
}

// ------------------------ WMMA fragment loaders ----------------------------
// A fragment 16x32 (MxK): lane (l&15) holds row m; K chunks {kof..kof+7},
// {kof+16..kof+23}, kof = 8*(l>=16).  (CDNA5 ISA)
__device__ __forceinline__ v16bf frag_a(const u16* A, int lda,
                                        int mBase, int k0, int lane) {
    const int m   = mBase + (lane & 15);
    const int kof = (lane >> 4) * 8;
    const u16* p  = A + (size_t)m * lda + k0 + kof;
    union { v16bf v; v8bf h[2]; } r;
    r.h[0] = *(const v8bf*)(p);
    r.h[1] = *(const v8bf*)(p + 16);
    return r.v;
}
// B fragment 32x16 (KxN); held as B^T row-major (N rows, K contig).
__device__ __forceinline__ v16bf frag_b(const u16* W, int ldw,
                                        int nBase, int k0, int lane) {
    const int n   = nBase + (lane & 15);
    const int kof = (lane >> 4) * 8;
    const u16* p  = W + (size_t)n * ldw + k0 + kof;
    union { v16bf v; v8bf h[2]; } r;
    r.h[0] = *(const v8bf*)(p);
    r.h[1] = *(const v8bf*)(p + 16);
    return r.v;
}

// ------------------------ TDM tile load (D# build) -------------------------
// DMA a [rows=128] x [cols=32] bf16 tile (row-major, row stride = ld elems)
// from global `gptr` into LDS at byte offset `lds_off` (contiguous 128x32).
// D# per CDNA5 ISA: group0 = {count/flags, lds_addr,
// global_addr[56:0], type=2}; group1 = {mask=0, data_size=2B, dims/strides}.
__device__ __forceinline__ void tdm_load_tile(const u16* gptr, unsigned lds_off,
                                              int ld /*elems*/) {
    const unsigned long long ga = (unsigned long long)(size_t)gptr;  // bytes
    v4u g0;
    g0[0] = 1u;                                        // count=1, user desc
    g0[1] = lds_off;                                   // lds_addr (bytes)
    g0[2] = (unsigned)(ga & 0xFFFFFFFFu);              // global_addr[31:0]
    g0[3] = (unsigned)((ga >> 32) & 0x01FFFFFFu)       // global_addr[56:32]
          | (2u << 30);                                // type = 2 ("image")
    v8i g1;
    g1[0] = (int)(1u << 16);                           // data_size=1 -> 2 bytes
    g1[1] = (int)(((unsigned)ld & 0xFFFFu) << 16);     // tensor_dim0[15:0]
    g1[2] = (int)((((unsigned)ld >> 16) & 0xFFFFu)     // tensor_dim0[31:16]
          | (128u << 16));                             // tensor_dim1[15:0]=128
    g1[3] = (int)(32u << 16);                          // tile_dim0 = 32
    g1[4] = (int)128u;                                 // tile_dim1 = 128
    g1[5] = (int)(unsigned)ld;                         // dim0_stride[31:0]
    g1[6] = 0;                                         // dim0_stride[47:32]
    g1[7] = 0;                                         // dim1_stride (2D)
    v4i g2 = {0, 0, 0, 0};
    v4i g3 = {0, 0, 0, 0};
#if __clang_major__ >= 23
    v8i g4 = {0, 0, 0, 0, 0, 0, 0, 0};
    __builtin_amdgcn_tensor_load_to_lds(g0, g1, g2, g3, g4, 0);
#else
    __builtin_amdgcn_tensor_load_to_lds(g0, g1, g2, g3, 0);
#endif
}

// ---------------- WMMA compute from a 128x32 LDS tile pair -----------------
// Wave tile 64x32: 4 A-frags x 2 B-frags -> 8 WMMAs per 32-wide K slice.
__device__ __forceinline__ void compute_slice(const u16* tA, const u16* tB,
                                              int mWave, int nWave, int lane,
                                              v8f acc[4][2]) {
    v16bf bf0, bf1;
    {
        const int n0  = nWave + (lane & 15);
        const int kof = (lane >> 4) * 8;
        union { v16bf v; v8bf h[2]; } r;
        const u16* p = tB + n0 * 32 + kof;
        r.h[0] = *(const v8bf*)(p);  r.h[1] = *(const v8bf*)(p + 16);
        bf0 = r.v;
        p += 16 * 32;
        r.h[0] = *(const v8bf*)(p);  r.h[1] = *(const v8bf*)(p + 16);
        bf1 = r.v;
    }
    #pragma unroll
    for (int i = 0; i < 4; ++i) {
        const int m   = mWave + 16 * i + (lane & 15);
        const int kof = (lane >> 4) * 8;
        union { v16bf v; v8bf h[2]; } r;
        const u16* p = tA + m * 32 + kof;
        r.h[0] = *(const v8bf*)(p);  r.h[1] = *(const v8bf*)(p + 16);
        acc[i][0] = wmma_bf16(r.v, bf0, acc[i][0]);
        acc[i][1] = wmma_bf16(r.v, bf1, acc[i][1]);
    }
}

__device__ __forceinline__ void store_tile(void* out, int MODE, int N,
                                           int mBlock, int nBlock,
                                           int mWave, int nWave, int lane,
                                           v8f acc[4][2]) {
    const int nl  = lane & 15;
    const int mro = (lane >> 4) * 8;
    #pragma unroll
    for (int i = 0; i < 4; ++i) {
        #pragma unroll
        for (int j = 0; j < 2; ++j) {
            const int n = nBlock + nWave + 16 * j + nl;
            #pragma unroll
            for (int r = 0; r < 8; ++r) {
                const int m = mBlock + mWave + 16 * i + mro + r;
                const size_t idx = (size_t)m * N + n;
                if (MODE == 0) ((float*)out)[idx] = acc[i][j][r];
                else           ((u16*)out)[idx]   = f2bf(acc[i][j][r]);
            }
        }
    }
}

// --------------- TDM-staged, double-buffered LDS GEMM (in_proj) ------------
// Block: 256 thr = 8 waves (2x4); block tile 128x128; wave tile 64x32.
// Wave 0 programs the TDM for slice kt+1 while all waves compute slice kt
// from the other LDS buffer (TENSORcnt + workgroup barrier for handoff).
template <int MODE>
__global__ void __launch_bounds__(256)
gemm_wmma_tdm(const u16* __restrict__ A, const u16* __restrict__ W,
              void* __restrict__ out, int M, int N, int K) {
    __shared__ __align__(16) u16 ldsA[2][128 * 32];   // 2 x 8 KB
    __shared__ __align__(16) u16 ldsB[2][128 * 32];   // 2 x 8 KB

    const int tid   = threadIdx.x;
    const int lane  = tid & 31;
    const int wave  = tid >> 5;
    const int mWave = (wave >> 2) * 64;
    const int nWave = (wave & 3) * 32;
    const int mBlock = blockIdx.y * 128;
    const int nBlock = blockIdx.x * 128;
    const bool loader = (wave == 0);

    v8f acc[4][2] = {};

    auto issue = [&](int kt, int buf) {
        const int k0 = kt << 5;
        tdm_load_tile(A + (size_t)mBlock * K + k0,
                      (unsigned)(size_t)(&ldsA[buf][0]), K);
        tdm_load_tile(W + (size_t)nBlock * K + k0,
                      (unsigned)(size_t)(&ldsB[buf][0]), K);
    };

    if (loader) issue(0, 0);
    const int KT = K >> 5;
    for (int kt = 0; kt < KT; ++kt) {
        if (loader) __builtin_amdgcn_s_wait_tensorcnt(0);  // slice kt landed
        __syncthreads();          // publish buf(kt&1); prior compute finished
        if (loader && kt + 1 < KT) issue(kt + 1, (kt + 1) & 1);
        compute_slice(ldsA[kt & 1], ldsB[kt & 1], mWave, nWave, lane, acc);
    }
    store_tile(out, MODE, N, mBlock, nBlock, mWave, nWave, lane, acc);
}

// ------------- register-staged LDS GEMM (out_proj, proven path) ------------
template <int MODE>
__global__ void __launch_bounds__(256)
gemm_wmma_lds(const u16* __restrict__ A, const u16* __restrict__ W,
              void* __restrict__ out, int M, int N, int K) {
    __shared__ __align__(16) u16 ldsA[128 * 32];
    __shared__ __align__(16) u16 ldsB[128 * 32];

    const int tid   = threadIdx.x;
    const int lane  = tid & 31;
    const int wave  = tid >> 5;
    const int mWave = (wave >> 2) * 64;
    const int nWave = (wave & 3) * 32;
    const int mBlock = blockIdx.y * 128;
    const int nBlock = blockIdx.x * 128;

    v8f acc[4][2] = {};
    v8bf aReg[2], bReg[2];

    auto stage_load = [&](int k0) {
        #pragma unroll
        for (int l = 0; l < 2; ++l) {
            const int idx = l * 2048 + tid * 8;
            const int row = idx >> 5;
            const int col = idx & 31;
            aReg[l] = *(const v8bf*)(A + (size_t)(mBlock + row) * K + k0 + col);
            bReg[l] = *(const v8bf*)(W + (size_t)(nBlock + row) * K + k0 + col);
        }
    };
    auto stage_store = [&]() {
        #pragma unroll
        for (int l = 0; l < 2; ++l) {
            const int idx = l * 2048 + tid * 8;
            *(v8bf*)(ldsA + idx) = aReg[l];
            *(v8bf*)(ldsB + idx) = bReg[l];
        }
    };

    stage_load(0);
    const int KT = K >> 5;
    for (int kt = 0; kt < KT; ++kt) {
        __syncthreads();
        stage_store();
        __syncthreads();
        if (kt + 1 < KT) stage_load((kt + 1) << 5);
        compute_slice(ldsA, ldsB, mWave, nWave, lane, acc);
    }
    store_tile(out, MODE, N, mBlock, nBlock, mWave, nWave, lane, acc);
}

// --------------------- direct WMMA GEMM (small GEMMs) ----------------------
// MODE 0: store fp32;  MODE 2: softplus(acc + bias[n]) fp32.
template <int MODE>
__global__ void __launch_bounds__(64)
gemm_wmma_direct(const u16* __restrict__ A, const u16* __restrict__ W,
                 void* __restrict__ out, const float* __restrict__ bias,
                 int M, int N, int K) {
    const int lane  = threadIdx.x & 31;
    const int wave  = threadIdx.x >> 5;
    const int nBase = blockIdx.x * 32;
    const int mBase = blockIdx.y * 64 + wave * 32;

    v8f acc00 = {}; v8f acc01 = {}; v8f acc10 = {}; v8f acc11 = {};
    for (int k0 = 0; k0 < K; k0 += 32) {
        v16bf a0 = frag_a(A, K, mBase,      k0, lane);
        v16bf a1 = frag_a(A, K, mBase + 16, k0, lane);
        v16bf b0 = frag_b(W, K, nBase,      k0, lane);
        v16bf b1 = frag_b(W, K, nBase + 16, k0, lane);
        acc00 = wmma_bf16(a0, b0, acc00);
        acc01 = wmma_bf16(a0, b1, acc01);
        acc10 = wmma_bf16(a1, b0, acc10);
        acc11 = wmma_bf16(a1, b1, acc11);
    }
    const int nl  = lane & 15;
    const int mro = (lane >> 4) * 8;
    #pragma unroll
    for (int i = 0; i < 2; ++i) {
        #pragma unroll
        for (int j = 0; j < 2; ++j) {
            v8f acc = (i == 0) ? ((j == 0) ? acc00 : acc01)
                               : ((j == 0) ? acc10 : acc11);
            const int n = nBase + 16 * j + nl;
            #pragma unroll
            for (int r = 0; r < 8; ++r) {
                const int m = mBase + 16 * i + mro + r;
                float v = acc[r];
                const size_t idx = (size_t)m * N + n;
                if (MODE == 0) {
                    ((float*)out)[idx] = v;
                } else {  // softplus(acc + bias[n])
                    v += bias[n];
                    v = (v > 20.0f) ? v : __logf(1.0f + __expf(v));
                    ((float*)out)[idx] = v;
                }
            }
        }
    }
}

// --------------------------- elementwise kernels ---------------------------
__global__ void f32_to_bf16_kernel(const float* __restrict__ in,
                                   u16* __restrict__ out, int n) {
    int i = blockIdx.x * blockDim.x + threadIdx.x;
    if (i < n) out[i] = f2bf(in[i]);
}

__global__ void dt_convert_kernel(const float* __restrict__ x_dbl,
                                  u16* __restrict__ dt_b) {
    int i = blockIdx.x * blockDim.x + threadIdx.x;      // NTOK*DTRANK
    if (i >= NTOK * DTRANK) return;
    int tok = i >> 6;
    int r   = i & 63;
    dt_b[i] = f2bf(x_dbl[(size_t)tok * (DTRANK + 2 * DSTATE) + r]);
}

// depthwise causal conv (k=4) + bias + SiLU; u = xz[:, :DINNER]
__global__ void conv_silu_kernel(const u16* __restrict__ xz_b,
                                 const float* __restrict__ conv_w,
                                 const float* __restrict__ conv_b,
                                 u16* __restrict__ u_c_b) {
    int idx = blockIdx.x * blockDim.x + threadIdx.x;    // NTOK*DINNER
    if (idx >= NTOK * DINNER) return;
    const int d   = idx & (DINNER - 1);
    const int tok = idx >> 11;
    const int t   = tok & (LL - 1);
    const int b   = tok >> 11;

    float acc = conv_b[d];
    #pragma unroll
    for (int j = 0; j < DCONV; ++j) {
        int tt = t - (DCONV - 1) + j;
        if (tt >= 0) {
            float uv = bf2f(xz_b[((size_t)(b * LL + tt)) * (2 * DINNER) + d]);
            acc += conv_w[d * DCONV + j] * uv;
        }
    }
    acc = acc / (1.0f + __expf(-acc));                  // SiLU
    u_c_b[idx] = f2bf(acc);
}

// -------------------------- selective-scan kernel --------------------------
// One lane per (b, d, s): 16 lanes cooperate on one channel's 16 states.
__global__ void __launch_bounds__(256)
scan_kernel(const float* __restrict__ delta,    // [NTOK][DINNER]
            const u16*   __restrict__ u_c_b,    // [NTOK][DINNER]
            const float* __restrict__ x_dbl,    // [NTOK][96]
            const u16*   __restrict__ xz_b,     // [NTOK][2*DINNER]
            const float* __restrict__ A_log,    // [DINNER][16]
            const float* __restrict__ Dp,       // [DINNER]
            u16*         __restrict__ y_b) {    // [NTOK][DINNER]
    const int g  = blockIdx.x * blockDim.x + threadIdx.x;  // BB*DINNER*16
    const int s  = g & 15;
    const int ch = g >> 4;
    const int b  = ch >> 11;
    const int d  = ch & (DINNER - 1);

    const float Areg = -__expf(A_log[(size_t)d * DSTATE + s]);
    const float Dd   = Dp[d];
    float h = 0.0f;

    for (int t = 0; t < LL; ++t) {
        const size_t tok = (size_t)b * LL + t;
        const float dt = delta[tok * DINNER + d];
        const float u  = bf2f(u_c_b[tok * DINNER + d]);
        const float Bs = x_dbl[tok * 96 + DTRANK + s];
        const float Cs = x_dbl[tok * 96 + DTRANK + DSTATE + s];
        h = __expf(dt * Areg) * h + (dt * u) * Bs;
        float y = h * Cs;
        y += __shfl_xor(y, 1);
        y += __shfl_xor(y, 2);
        y += __shfl_xor(y, 4);
        y += __shfl_xor(y, 8);                 // sum over 16 states
        if (s == 0) {
            y += Dd * u;
            const float z = bf2f(xz_b[tok * (2 * DINNER) + DINNER + d]);
            y *= z / (1.0f + __expf(-z));      // gate with SiLU(z)
            y_b[tok * DINNER + d] = f2bf(y);
        }
    }
}

// ------------------------------- launcher ----------------------------------
extern "C" void kernel_launch(void* const* d_in, const int* in_sizes, int n_in,
                              void* d_out, int out_size, void* d_ws, size_t ws_size,
                              hipStream_t stream) {
    (void)in_sizes; (void)n_in; (void)out_size; (void)ws_size;

    const float* x        = (const float*)d_in[0];
    const float* W_in     = (const float*)d_in[1];
    const float* conv_w   = (const float*)d_in[2];
    const float* conv_b   = (const float*)d_in[3];
    const float* W_xproj  = (const float*)d_in[4];
    const float* W_dtproj = (const float*)d_in[5];
    const float* dt_bias  = (const float*)d_in[6];
    const float* A_log    = (const float*)d_in[7];
    const float* Dp       = (const float*)d_in[8];
    const float* W_out    = (const float*)d_in[9];
    float* out            = (float*)d_out;

    // ---- workspace layout (~119 MB) ----
    char* ws = (char*)d_ws;
    size_t off = 0;
    auto alloc = [&](size_t bytes) -> char* {
        char* p = ws + off;
        off += (bytes + 255) & ~(size_t)255;
        return p;
    };
    u16*   xb        = (u16*)  alloc((size_t)NTOK * DMODEL * 2);
    u16*   W_in_b    = (u16*)  alloc((size_t)2 * DINNER * DMODEL * 2);
    u16*   W_xproj_b = (u16*)  alloc((size_t)96 * DINNER * 2);
    u16*   W_dtproj_b= (u16*)  alloc((size_t)DINNER * DTRANK * 2);
    u16*   W_out_b   = (u16*)  alloc((size_t)DMODEL * DINNER * 2);
    u16*   xz_b      = (u16*)  alloc((size_t)NTOK * 2 * DINNER * 2);
    u16*   u_c_b     = (u16*)  alloc((size_t)NTOK * DINNER * 2);
    float* x_dbl     = (float*)alloc((size_t)NTOK * 96 * 4);
    u16*   dt_b      = (u16*)  alloc((size_t)NTOK * DTRANK * 2);
    float* delta     = (float*)alloc((size_t)NTOK * DINNER * 4);
    u16*   y_b       = (u16*)  alloc((size_t)NTOK * DINNER * 2);

    const int CT = 256;
    auto cvt = [&](const float* src, u16* dst, int n) {
        f32_to_bf16_kernel<<<(n + CT - 1) / CT, CT, 0, stream>>>(src, dst, n);
    };

    // 1) fp32 -> bf16 operand conversion
    cvt(x,        xb,         NTOK * DMODEL);
    cvt(W_in,     W_in_b,     2 * DINNER * DMODEL);
    cvt(W_xproj,  W_xproj_b,  96 * DINNER);
    cvt(W_dtproj, W_dtproj_b, DINNER * DTRANK);
    cvt(W_out,    W_out_b,    DMODEL * DINNER);

    // 2) in_proj: xz = x @ W_in^T   (bf16 out, TDM-staged LDS tiles)
    gemm_wmma_tdm<1><<<dim3(2 * DINNER / 128, NTOK / 128), 256, 0, stream>>>(
        xb, W_in_b, xz_b, NTOK, 2 * DINNER, DMODEL);

    // 3) depthwise causal conv + SiLU -> u_c (bf16)
    {
        int n = NTOK * DINNER;
        conv_silu_kernel<<<(n + CT - 1) / CT, CT, 0, stream>>>(
            xz_b, conv_w, conv_b, u_c_b);
    }

    // 4) x_proj: x_dbl = u_c @ W_xproj^T   (fp32 out, direct)
    gemm_wmma_direct<0><<<dim3(96 / 32, NTOK / 64), 64, 0, stream>>>(
        u_c_b, W_xproj_b, x_dbl, nullptr, NTOK, 96, DINNER);

    // 5) dt slice -> bf16
    {
        int n = NTOK * DTRANK;
        dt_convert_kernel<<<(n + CT - 1) / CT, CT, 0, stream>>>(x_dbl, dt_b);
    }

    // 6) delta = softplus(dt @ W_dtproj^T + dt_bias)   (fp32 out, direct)
    gemm_wmma_direct<2><<<dim3(DINNER / 32, NTOK / 64), 64, 0, stream>>>(
        dt_b, W_dtproj_b, delta, dt_bias, NTOK, DINNER, DTRANK);

    // 7) selective scan + skip + SiLU(z) gate -> y (bf16), 16 lanes/channel
    {
        int n = BB * DINNER * DSTATE;   // 65536
        scan_kernel<<<n / CT, CT, 0, stream>>>(
            delta, u_c_b, x_dbl, xz_b, A_log, Dp, y_b);
    }

    // 8) out_proj: out = y @ W_out^T   (fp32 out, register-staged LDS tiles)
    gemm_wmma_lds<0><<<dim3(DMODEL / 128, NTOK / 128), 256, 0, stream>>>(
        y_b, W_out_b, out, NTOK, DMODEL, DINNER);
}
